// SelfAttention_77360950935608
// MI455X (gfx1250) — matrix operand order repaired and checked by reading, not compiled, not run on hardware
//
#include <hip/hip_runtime.h>
#include <hip/hip_bf16.h>

#define EMB   768
#define SEQ   2048
#define NB    2
#define HEADS 12
#define HDIM  64

typedef __attribute__((ext_vector_type(16))) _Float16 v16h;
typedef __attribute__((ext_vector_type(8)))  _Float16 v8h;
typedef __attribute__((ext_vector_type(8)))  float    v8f;

static __device__ inline v16h mk16(v8h a, v8h b) {
  v16h r;
#pragma unroll
  for (int i = 0; i < 8; ++i) { r[i] = a[i]; r[i + 8] = b[i]; }
  return r;
}

static __device__ inline v8f wmma_f16(v16h a, v16h b, v8f c) {
  // D = A(16x32 f16) * B(32x16 f16) + C(16x16 f32)
  return __builtin_amdgcn_wmma_f32_16x16x32_f16(false, a, false, b,
                                                (short)0, c, false, false);
}

// Generic->LDS(AS3) pointer, 32-bit LDS byte offset for async-DMA operands.
static __device__ inline uint32_t lds_off(const void* p) {
  return (uint32_t)(uintptr_t)(__attribute__((address_space(3))) const void*)p;
}

// CDNA5 async global->LDS copy (16B per lane), tracked by ASYNCcnt.
static __device__ inline void async_ld_b128(uint32_t lds, const void* gptr) {
  asm volatile("global_load_async_to_lds_b128 %0, %1, off"
               :: "v"(lds), "v"(gptr) : "memory");
}
static __device__ inline void wait_async0() {
  asm volatile("s_wait_asynccnt 0x0" ::: "memory");
}
static __device__ inline void wait_async2() {  // allow next chunk in flight
  asm volatile("s_wait_asynccnt 0x2" ::: "memory");
}

// ---------------------------------------------------------------------------
// GEMM: out[M x EMB] = (A[M x EMB] @ W^T + bias) * scale   ("TN" layout)
// Block: 256 threads = 8 waves; tile 128(M) x 64(N); K staged in LDS by 64.
// ---------------------------------------------------------------------------
template <typename TA, bool HALF_OUT>
__global__ __launch_bounds__(256) void gemm_bias_kernel(
    const TA* __restrict__ A, const float* __restrict__ W,
    const float* __restrict__ bias, void* __restrict__ outp, float scale) {
  __shared__ _Float16 As[128][64];  // 16 KB (reused for output staging)
  __shared__ _Float16 Bs[64][64];   //  8 KB

  const int t = threadIdx.x;
  const int w = t >> 5, lane = t & 31, hi = lane >> 4, ln = lane & 15;
  const int m0 = blockIdx.x * 128;
  const int n0 = blockIdx.y * 64;

  v8f acc[4] = {};

  const int ar = t >> 1, ac = (t & 1) * 32;  // A: 128x64, 32 elems/thread
  const int br = t >> 2, bc = (t & 3) * 16;  // B: 64x64, 16 elems/thread

  for (int k = 0; k < EMB; k += 64) {
    __syncthreads();
    const TA* ap = A + (size_t)(m0 + ar) * EMB + k + ac;
#pragma unroll
    for (int i = 0; i < 32; ++i) As[ar][ac + i] = (_Float16)ap[i];
    if (k + 64 < EMB) __builtin_prefetch(ap + 64, 0, 3);
    const float* bp = W + (size_t)(n0 + br) * EMB + k + bc;
#pragma unroll
    for (int i = 0; i < 16; ++i) Bs[br][bc + i] = (_Float16)bp[i];
    __syncthreads();

    const int arow = w * 16 + ln;
    v16h af[2];
#pragma unroll
    for (int kk = 0; kk < 2; ++kk)
      af[kk] = mk16(*(const v8h*)&As[arow][kk * 32 + hi * 8],
                    *(const v8h*)&As[arow][kk * 32 + 16 + hi * 8]);
#pragma unroll
    for (int nt = 0; nt < 4; ++nt)
#pragma unroll
      for (int kk = 0; kk < 2; ++kk) {
        v16h bf = mk16(*(const v8h*)&Bs[nt * 16 + ln][kk * 32 + hi * 8],
                       *(const v8h*)&Bs[nt * 16 + ln][kk * 32 + 16 + hi * 8]);
        acc[nt] = wmma_f16(af[kk], bf, acc[nt]);
      }
  }

  // Epilogue: C layout -> VGPR j holds M = j + 8*hi, N = ln
  if constexpr (HALF_OUT) {
    // stage through LDS (reuse As) for coalesced b128 global stores
    __syncthreads();
#pragma unroll
    for (int nt = 0; nt < 4; ++nt) {
      const float bv = bias[n0 + nt * 16 + ln];
#pragma unroll
      for (int j = 0; j < 8; ++j)
        As[w * 16 + j + hi * 8][nt * 16 + ln] =
            (_Float16)((acc[nt][j] + bv) * scale);
    }
    __syncthreads();
    const int orow = t >> 1, ocol = (t & 1) * 32;  // 64B per thread
    _Float16* op = (_Float16*)outp + (size_t)(m0 + orow) * EMB + n0 + ocol;
#pragma unroll
    for (int i = 0; i < 4; ++i)
      *(v8h*)(op + i * 8) = *(const v8h*)&As[orow][ocol + i * 8];
  } else {
#pragma unroll
    for (int nt = 0; nt < 4; ++nt) {
      const int col = n0 + nt * 16 + ln;
      const float bv = bias[col];
#pragma unroll
      for (int j = 0; j < 8; ++j) {
        const int row = m0 + w * 16 + j + hi * 8;
        ((float*)outp)[(size_t)row * EMB + col] = (acc[nt][j] + bv) * scale;
      }
    }
  }
}

// ---------------------------------------------------------------------------
// Causal flash attention over f16 q/k/v (q pre-scaled by 1/8).
// Block = (q-tile of 128, head, batch); 8 waves x 16-row Q strip each.
// 64-key chunks, double-buffered: K tiles via async global->LDS DMA overlap
// the previous chunk's 16 WMMAs; V staged transposed through registers.
// ---------------------------------------------------------------------------
__global__ __launch_bounds__(256) void attn_kernel(
    const _Float16* __restrict__ Q, const _Float16* __restrict__ Kv,
    const _Float16* __restrict__ V, _Float16* __restrict__ O) {
  __shared__ _Float16 kt[2][64][64];   // 16 KB, keys x dims (double-buffered)
  __shared__ _Float16 vt[2][64][64];   // 16 KB, dims x keys (transposed)
  __shared__ _Float16 ps[8][16][64];   // 16 KB, per-wave P / O staging

  const int t = threadIdx.x;
  const int w = t >> 5, lane = t & 31, hi = lane >> 4, ln = lane & 15;
  const int b = blockIdx.z, h = blockIdx.y, q0 = blockIdx.x * 128;
  const int qw = q0 + w * 16;

  // staging mapping: 512 x 16B chunks per tile, 2 per thread
  const int key_ld = t >> 3;            // 0..31  (rep adds +32)
  const int dim_ld = (t & 7) * 8;       // 0..56
  const size_t base = (size_t)b * SEQ * EMB + h * HDIM;

  // Q fragments resident in registers: 16 rows x 64 dims -> two 16x32 A-frags
  const _Float16* qp = Q + base + (size_t)(qw + ln) * EMB;
  v16h qa[2];
#pragma unroll
  for (int kk = 0; kk < 2; ++kk)
    qa[kk] = mk16(*(const v8h*)(qp + kk * 32 + hi * 8),
                  *(const v8h*)(qp + kk * 32 + 16 + hi * 8));

  v8f oa[4] = {};
  float mrow[8], lrow[8];
#pragma unroll
  for (int j = 0; j < 8; ++j) { mrow[j] = -1.0e30f; lrow[j] = 0.0f; }

  const int nchunk = q0 / 64 + 2;  // causal: keys <= last q row of block

  // ---- prologue: issue chunk 0 K-DMA and V register loads
  v8h va[2];
#pragma unroll
  for (int rep = 0; rep < 2; ++rep) {
    const int key = key_ld + rep * 32;
    const size_t roff = base + (size_t)key * EMB + dim_ld;
    async_ld_b128(lds_off(&kt[0][key][dim_ld]), Kv + roff);
    va[rep] = *(const v8h*)(V + roff);
  }

  for (int c = 0; c < nchunk; ++c) {
    const int k0 = c * 64;
    const int buf = c & 1;
    __syncthreads();  // kt[buf^1] / vt[buf] readers from earlier chunks done

    // commit V(c) to LDS transposed
#pragma unroll
    for (int rep = 0; rep < 2; ++rep) {
      const int key = key_ld + rep * 32;
#pragma unroll
      for (int i = 0; i < 8; ++i) vt[buf][dim_ld + i][key] = va[rep][i];
    }

    if (c + 1 < nchunk) {  // prefetch chunk c+1 while chunk c computes
      const int nk0 = k0 + 64;
#pragma unroll
      for (int rep = 0; rep < 2; ++rep) {
        const int key = key_ld + rep * 32;
        const size_t roff = base + (size_t)(nk0 + key) * EMB + dim_ld;
        async_ld_b128(lds_off(&kt[buf ^ 1][key][dim_ld]), Kv + roff);
        va[rep] = *(const v8h*)(V + roff);
      }
      wait_async2();  // in-order: chunk c's 2 DMAs done, next 2 in flight
    } else {
      wait_async0();
    }
    __syncthreads();  // tile (c) published to all waves

    if (k0 <= qw + 15) {  // chunk touches this wave's causal range
      // S = Q @ K^T : four 16x16 key tiles, head-dim reduced as 2 x K32
      v8f s[4] = {};
#pragma unroll
      for (int nt = 0; nt < 4; ++nt)
#pragma unroll
        for (int kk = 0; kk < 2; ++kk) {
          v16h bf = mk16(*(const v8h*)&kt[buf][nt * 16 + ln][kk * 32 + hi * 8],
                         *(const v8h*)&kt[buf][nt * 16 + ln][kk * 32 + 16 + hi * 8]);
          s[nt] = wmma_f16(qa[kk], bf, s[nt]);
        }

      // causal mask: only the wave's diagonal chunk needs it
      if (k0 + 63 > qw) {
#pragma unroll
        for (int nt = 0; nt < 4; ++nt) {
          const int key = k0 + nt * 16 + ln;
#pragma unroll
          for (int j = 0; j < 8; ++j)
            if (key > qw + j + hi * 8) s[nt][j] = -1.0e30f;
        }
      }

      // online softmax: row lives in one 16-lane half-group per VGPR j
#pragma unroll
      for (int j = 0; j < 8; ++j) {
        float mx = fmaxf(fmaxf(s[0][j], s[1][j]), fmaxf(s[2][j], s[3][j]));
#pragma unroll
        for (int off = 8; off > 0; off >>= 1)
          mx = fmaxf(mx, __shfl_xor(mx, off, 32));
        const float mnew = fmaxf(mrow[j], mx);
        float sum = 0.0f;
#pragma unroll
        for (int nt = 0; nt < 4; ++nt) {
          const float p = __expf(s[nt][j] - mnew);
          s[nt][j] = p;
          sum += p;
        }
#pragma unroll
        for (int off = 8; off > 0; off >>= 1)
          sum += __shfl_xor(sum, off, 32);
        const float corr = __expf(mrow[j] - mnew);
        lrow[j] = lrow[j] * corr + sum;
        mrow[j] = mnew;
#pragma unroll
        for (int nt = 0; nt < 4; ++nt) oa[nt][j] *= corr;
      }

      // re-layout P: C-fragment -> row-major 16x64 in wave-private LDS
#pragma unroll
      for (int nt = 0; nt < 4; ++nt)
#pragma unroll
        for (int j = 0; j < 8; ++j)
          ps[w][j + hi * 8][nt * 16 + ln] = (_Float16)s[nt][j];

      v16h pa[2];
#pragma unroll
      for (int kk = 0; kk < 2; ++kk)
        pa[kk] = mk16(*(const v8h*)&ps[w][ln][kk * 32 + hi * 8],
                      *(const v8h*)&ps[w][ln][kk * 32 + 16 + hi * 8]);

      // O += P @ V : 4 head-dim column tiles, key dim reduced as 2 x K32
#pragma unroll
      for (int nt = 0; nt < 4; ++nt)
#pragma unroll
        for (int kk = 0; kk < 2; ++kk) {
          v16h bv = mk16(*(const v8h*)&vt[buf][nt * 16 + ln][kk * 32 + hi * 8],
                         *(const v8h*)&vt[buf][nt * 16 + ln][kk * 32 + 16 + hi * 8]);
          oa[nt] = wmma_f16(pa[kk], bv, oa[nt]);
        }
    }
  }

  // normalize; stage O through wave-private LDS for coalesced b128 stores
#pragma unroll
  for (int j = 0; j < 8; ++j) {
    const float inv = 1.0f / lrow[j];
#pragma unroll
    for (int nt = 0; nt < 4; ++nt)
      ps[w][j + hi * 8][nt * 16 + ln] = (_Float16)(oa[nt][j] * inv);
  }
  // 16 rows x 128B per wave -> 4 x 16B per lane
#pragma unroll
  for (int i = 0; i < 4; ++i) {
    const int cc = lane + 32 * i;           // 0..127
    const int row = cc >> 3, off = (cc & 7) * 8;
    _Float16* op = O + base + (size_t)(qw + row) * EMB + off;
    *(v8h*)op = *(const v8h*)&ps[w][row][off];
  }
}

// ---------------------------------------------------------------------------
extern "C" void kernel_launch(void* const* d_in, const int* in_sizes, int n_in,
                              void* d_out, int out_size, void* d_ws,
                              size_t ws_size, hipStream_t stream) {
  (void)in_sizes; (void)n_in; (void)out_size; (void)ws_size;
  const float* x    = (const float*)d_in[0];
  const float* wq_w = (const float*)d_in[1];
  const float* wq_b = (const float*)d_in[2];
  const float* wk_w = (const float*)d_in[3];
  const float* wk_b = (const float*)d_in[4];
  const float* wv_w = (const float*)d_in[5];
  const float* wv_b = (const float*)d_in[6];
  const float* wo_w = (const float*)d_in[7];
  const float* wo_b = (const float*)d_in[8];

  const int M = NB * SEQ;  // 4096 rows
  _Float16* qb = (_Float16*)d_ws;
  _Float16* kb = qb + (size_t)M * EMB;
  _Float16* vb = kb + (size_t)M * EMB;
  _Float16* ob = vb + (size_t)M * EMB;   // total f16 workspace: 24 MB

  dim3 ggrid(M / 128, EMB / 64);
  // q/k/v projections (q carries the 1/sqrt(head_dim)=1/8 scale)
  gemm_bias_kernel<float, true><<<ggrid, 256, 0, stream>>>(x, wq_w, wq_b, qb, 0.125f);
  gemm_bias_kernel<float, true><<<ggrid, 256, 0, stream>>>(x, wk_w, wk_b, kb, 1.0f);
  gemm_bias_kernel<float, true><<<ggrid, 256, 0, stream>>>(x, wv_w, wv_b, vb, 1.0f);

  attn_kernel<<<dim3(SEQ / 128, HEADS, NB), 256, 0, stream>>>(qb, kb, vb, ob);

  // output projection -> f32 result
  gemm_bias_kernel<_Float16, false><<<ggrid, 256, 0, stream>>>(ob, wo_w, wo_b,
                                                               d_out, 1.0f);
}
